// VectorQuantizer_30648886624776
// MI455X (gfx1250) — matrix operand order repaired and checked by reading, not compiled, not run on hardware
//
#include <hip/hip_runtime.h>
#include <hip/hip_bf16.h>

typedef __attribute__((ext_vector_type(16))) _Float16 v16h;
typedef __attribute__((ext_vector_type(8)))  _Float16 v8h;
typedef __attribute__((ext_vector_type(8)))  float    v8f;

#define N_TOTAL   131072      // 8*16384 rows
#define CODE_DIM  64
#define N_CODES   1024
#define DECAY     0.99f
#define EPSILON   1e-5f

// ---- workspace layout (bytes) ----
// emb16 and enorm are contiguous so one async-staging pass covers both.
#define WS_EMB16   0          // 1024*64 f16          = 131072 B
#define WS_ENORM   131072     // 1024 f32             =   4096 B
#define WS_STAGE_B 135168     // bytes staged to LDS (emb16 + enorm)
#define WS_CNT     135168     // 1024 f32             =   4096 B
#define WS_DW      139264     // 1024*64 f32          = 262144 B
#define WS_LOSS    401408     // 1 f32
#define WS_ZERO_N  66561      // floats from WS_CNT..WS_LOSS inclusive

// ---- d_out layout (floats) ----
#define O_ZQ    0            // 8388608
#define O_IDX   8388608      // 131072
#define O_LOSS  8519680      // 1
#define O_PERP  8519681      // 1
#define O_EMB   8519682      // 65536
#define O_ECS   8585218      // 1024
#define O_EMAW  8586242      // 65536

__global__ __launch_bounds__(256) void vq_zero(float* p, int n) {
  int i = blockIdx.x * 256 + threadIdx.x;
  if (i < n) p[i] = 0.0f;
}

// one thread per code: f32 codebook -> f16 copy + ||e||^2
__global__ __launch_bounds__(256) void vq_prep(const float* __restrict__ emb32,
                                               _Float16* __restrict__ emb16,
                                               float* __restrict__ enorm) {
  int t = blockIdx.x * 256 + threadIdx.x;
  if (t < N_CODES) {
    float s = 0.0f;
    #pragma unroll 8
    for (int d = 0; d < CODE_DIM; ++d) {
      float v = emb32[t * CODE_DIM + d];
      s += v * v;
      emb16[t * CODE_DIM + d] = (_Float16)v;
    }
    enorm[t] = s;
  }
}

__global__ __launch_bounds__(256) void vq_main(const float* __restrict__ z,
                                               const float* __restrict__ emb32,
                                               const _Float16* __restrict__ stage_src, // ws base (emb16|enorm)
                                               float* __restrict__ out_zq,
                                               float* __restrict__ out_idx,
                                               float* __restrict__ cnt,
                                               float* __restrict__ dw,
                                               float* __restrict__ loss_acc) {
  // single LDS buffer: [0,131072) f16 codebook, [131072,135168) ||e||^2,
  // [135168, +512) per-wave index exchange.  135680 B of 320 KB/WGP.
  __shared__ __align__(16) unsigned char s_buf[WS_STAGE_B + 8 * 16 * 4];
  _Float16* s_emb   = (_Float16*)s_buf;
  float*    s_enorm = (float*)(s_buf + WS_ENORM);
  int*      s_idx   = (int*)(s_buf + WS_STAGE_B);

  const int t = threadIdx.x;

  // ---- async global->LDS staging of f16 codebook + norms (no VGPR bounce) ----
  // GLOBAL_LOAD_ASYNC_TO_LDS_B128, GVS mode: mem = SADDR + VADDR(32b offset);
  // LDS[vdst + 0] written; tracked by ASYNCcnt.
  {
    const unsigned int lds_base = (unsigned int)(size_t)s_buf;  // wg-relative
    const unsigned long long gbase = (unsigned long long)(size_t)stage_src;
    #pragma unroll 3
    for (int i = t; i < WS_STAGE_B / 16; i += 256) {
      unsigned int ldsa = lds_base + (unsigned int)(i * 16);
      unsigned int goff = (unsigned int)(i * 16);
      asm volatile("global_load_async_to_lds_b128 %0, %1, %2"
                   :: "v"(ldsa), "v"(goff), "s"(gbase)
                   : "memory");
    }
    asm volatile("s_wait_asynccnt 0x0" ::: "memory");
  }
  __syncthreads();

  const int wv   = t >> 5;
  const int lane = t & 31;
  const int col  = lane & 15;     // N-column / M-row-in-tile for A
  const int hf   = lane >> 4;     // half-wave select
  const int tile = blockIdx.x * 8 + wv;
  const int rowg = tile * 16 + col;

  // ---- load z row, build A fragments (ISA 16-bit A 16x32 layout) ----
  // lane (hf,col): run r4 covers K = r4*16 + hf*8 .. +7
  float zr[32];
  const float* zp = z + (size_t)rowg * CODE_DIM;
  #pragma unroll
  for (int r4 = 0; r4 < 4; ++r4) {
    const int K0 = r4 * 16 + hf * 8;
    float4 p0 = *(const float4*)(zp + K0);
    float4 p1 = *(const float4*)(zp + K0 + 4);
    zr[r4*8+0]=p0.x; zr[r4*8+1]=p0.y; zr[r4*8+2]=p0.z; zr[r4*8+3]=p0.w;
    zr[r4*8+4]=p1.x; zr[r4*8+5]=p1.y; zr[r4*8+6]=p1.z; zr[r4*8+7]=p1.w;
  }
  v16h a0, a1;
  #pragma unroll
  for (int j = 0; j < 8; ++j) {
    a0[j]     = (_Float16)zr[j];        // K 0..31 : elems 0..7  = K hf*8+j
    a0[8 + j] = (_Float16)zr[8 + j];    //           elems 8..15 = K 16+hf*8+j
    a1[j]     = (_Float16)zr[16 + j];   // K 32..63
    a1[8 + j] = (_Float16)zr[24 + j];
  }

  // ---- distance argmin over 64 code tiles via WMMA ----
  float bestv[8];
  int   besti[8];
  #pragma unroll
  for (int r = 0; r < 8; ++r) { bestv[r] = 3.4e38f; besti[r] = 0; }

  for (int c = 0; c < 64; ++c) {
    const int code = c * 16 + col;
    // B[k][n] = emb16[code(n)][k]; elems 0..7 = K hf*8+j, 8..15 = K 16+hf*8+j (two 16B ds loads)
    const v16h b0 = __builtin_shufflevector(*(const v8h*)(s_emb + code * CODE_DIM + hf * 8), *(const v8h*)(s_emb + code * CODE_DIM + 16 + hf * 8), 0, 1, 2, 3, 4, 5, 6, 7, 8, 9, 10, 11, 12, 13, 14, 15);
    const v16h b1 = __builtin_shufflevector(*(const v8h*)(s_emb + code * CODE_DIM + 32 + hf * 8), *(const v8h*)(s_emb + code * CODE_DIM + 48 + hf * 8), 0, 1, 2, 3, 4, 5, 6, 7, 8, 9, 10, 11, 12, 13, 14, 15);
    v8f acc = {};
    acc = __builtin_amdgcn_wmma_f32_16x16x32_f16(false, a0, false, b0, (short)0, acc, false, false);
    acc = __builtin_amdgcn_wmma_f32_16x16x32_f16(false, a1, false, b1, (short)0, acc, false, false);
    const float en = s_enorm[code];
    #pragma unroll
    for (int r = 0; r < 8; ++r) {
      float d = en - 2.0f * acc[r];   // ||z||^2 constant per row -> drop
      if (d < bestv[r]) { bestv[r] = d; besti[r] = code; }
    }
  }

  // ---- cross-lane min-reduce over the 16 columns (per half-wave) ----
  #pragma unroll
  for (int m = 1; m <= 8; m <<= 1) {
    #pragma unroll
    for (int r = 0; r < 8; ++r) {
      float ov = __shfl_xor(bestv[r], m, 32);
      int   oi = __shfl_xor(besti[r], m, 32);
      if (ov < bestv[r] || (ov == bestv[r] && oi < besti[r])) {
        bestv[r] = ov; besti[r] = oi;
      }
    }
  }
  // C layout: lanes 0-15 hold rows 0..7 (vgpr r), lanes 16-31 rows 8..15
  if (col == 0) {
    #pragma unroll
    for (int r = 0; r < 8; ++r) s_idx[wv * 16 + hf * 8 + r] = besti[r];
  }
  __builtin_amdgcn_wave_barrier();   // keep ds store->load ordered (same wave)

  const int myidx = s_idx[wv * 16 + col];

  if (lane < 16) {
    int ci = s_idx[wv * 16 + lane];
    out_idx[tile * 16 + lane] = (float)ci;
    atomicAdd(&cnt[ci], 1.0f);
  }

  // ---- gather z_q (fp32 codebook), loss, dw scatter ----
  const float* ep  = emb32 + (size_t)myidx * CODE_DIM;
  float*       op  = out_zq + (size_t)rowg * CODE_DIM;
  float*       dwp = dw + (size_t)myidx * CODE_DIM;
  float sq = 0.0f;
  #pragma unroll
  for (int r4 = 0; r4 < 4; ++r4) {
    const int K0 = r4 * 16 + hf * 8;
    float4 e0 = *(const float4*)(ep + K0);
    float4 e1 = *(const float4*)(ep + K0 + 4);
    *(float4*)(op + K0)     = e0;        // z_q_st == z_q numerically
    *(float4*)(op + K0 + 4) = e1;
    float d;
    d = e0.x - zr[r4*8+0]; sq += d*d;  d = e0.y - zr[r4*8+1]; sq += d*d;
    d = e0.z - zr[r4*8+2]; sq += d*d;  d = e0.w - zr[r4*8+3]; sq += d*d;
    d = e1.x - zr[r4*8+4]; sq += d*d;  d = e1.y - zr[r4*8+5]; sq += d*d;
    d = e1.z - zr[r4*8+6]; sq += d*d;  d = e1.w - zr[r4*8+7]; sq += d*d;
    #pragma unroll
    for (int j = 0; j < 8; ++j) atomicAdd(dwp + K0 + j, zr[r4*8 + j]);
  }
  #pragma unroll
  for (int m = 1; m <= 16; m <<= 1) sq += __shfl_xor(sq, m, 32);
  if (lane == 0) atomicAdd(loss_acc, sq);
}

__device__ __forceinline__ float block_sum_1024(float v, float* red) {
  const int t = threadIdx.x;
  __syncthreads();
  red[t] = v;
  __syncthreads();
  #pragma unroll
  for (int s = 512; s > 0; s >>= 1) {
    if (t < s) red[t] += red[t + s];
    __syncthreads();
  }
  float r = red[0];
  __syncthreads();
  return r;
}

__global__ __launch_bounds__(1024) void vq_finalize(const float* __restrict__ ecs_old,
                                                    const float* __restrict__ emaw_old,
                                                    const float* __restrict__ cnt,
                                                    const float* __restrict__ dw,
                                                    const float* __restrict__ loss_acc,
                                                    float* __restrict__ out_loss,
                                                    float* __restrict__ out_perp,
                                                    float* __restrict__ out_emb,
                                                    float* __restrict__ out_ecs,
                                                    float* __restrict__ out_emaw) {
  __shared__ float red[1024];
  const int t = threadIdx.x;

  const float c    = cnt[t];
  const float necs = ecs_old[t] * DECAY + (1.0f - DECAY) * c;
  out_ecs[t] = necs;

  const float ntot  = block_sum_1024(necs, red);
  const float total = block_sum_1024(c, red);

  float p0 = fmaxf(c / total, 1e-10f);
  const float s2 = block_sum_1024(p0, red);
  float p  = p0 / s2;
  const float h = block_sum_1024(p * logf(p), red);

  float cs = (necs + EPSILON) / (ntot + (float)N_CODES * EPSILON) * ntot;
  cs = fmaxf(cs, 0.01f);

  #pragma unroll 8
  for (int d = 0; d < CODE_DIM; ++d) {
    const float nw = emaw_old[t * CODE_DIM + d] * DECAY + (1.0f - DECAY) * dw[t * CODE_DIM + d];
    out_emaw[t * CODE_DIM + d] = nw;
    out_emb[t * CODE_DIM + d]  = nw / cs;
  }

  if (t == 0) {
    // commitment and codebook losses are numerically identical (stop-grad only
    // affects gradients): loss = (0.5 + 1.0) * mean((z_q - z)^2)
    out_loss[0] = 1.5f * loss_acc[0] / (float)((size_t)N_TOTAL * CODE_DIM);
    out_perp[0] = expf(-h);
  }
}

extern "C" void kernel_launch(void* const* d_in, const int* in_sizes, int n_in,
                              void* d_out, int out_size, void* d_ws, size_t ws_size,
                              hipStream_t stream) {
  const float* z      = (const float*)d_in[0];   // [8,16384,64]
  const float* emb32  = (const float*)d_in[1];   // [1024,64]
  const float* ecs    = (const float*)d_in[2];   // [1024]
  const float* emaw   = (const float*)d_in[3];   // [1024,64]

  char* ws = (char*)d_ws;
  _Float16* emb16   = (_Float16*)(ws + WS_EMB16);
  float*    enorm   = (float*)(ws + WS_ENORM);
  float*    cnt     = (float*)(ws + WS_CNT);
  float*    dw      = (float*)(ws + WS_DW);
  float*    lossacc = (float*)(ws + WS_LOSS);

  float* out = (float*)d_out;
  float* out_zq   = out + O_ZQ;
  float* out_idx  = out + O_IDX;
  float* out_loss = out + O_LOSS;
  float* out_perp = out + O_PERP;
  float* out_emb  = out + O_EMB;
  float* out_ecs  = out + O_ECS;
  float* out_emaw = out + O_EMAW;

  vq_zero<<<(WS_ZERO_N + 255) / 256, 256, 0, stream>>>((float*)(ws + WS_CNT), WS_ZERO_N);
  vq_prep<<<(N_CODES + 255) / 256, 256, 0, stream>>>(emb32, emb16, enorm);
  vq_main<<<N_TOTAL / 128, 256, 0, stream>>>(z, emb32, (const _Float16*)ws,
                                             out_zq, out_idx, cnt, dw, lossacc);
  vq_finalize<<<1, 1024, 0, stream>>>(ecs, emaw, cnt, dw, lossacc,
                                      out_loss, out_perp, out_emb, out_ecs, out_emaw);
}
